// SpexphormerAttention_79774722556001
// MI455X (gfx1250) — matrix operand order repaired and checked by reading, not compiled, not run on hardware
//
#include <hip/hip_runtime.h>
#include <hip/hip_bf16.h>

// Spexphormer graph attention, MI455X (gfx1250), wave32 + WMMA f16.
//
// Shapes (fixed by the harness):
//   N1 = N2 = 32768, DEG = 16, IN = 128, H = 8, D = 16, H*D = 128
//
// Roofline: ~21.5 GFLOP vs ~0.9 GB HBM traffic -> memory bound at 23.3 TB/s
// (~38 us). Strategy: keep e1 (would be 268 MB) entirely on-chip, run all
// GEMM math through v_wmma_f32_16x16x32_f16, stream edge_attr once.

#define N1C   32768
#define N2C   32768
#define INDIM 128
#define HDIM  128   // H*D
#define HEADS 8
#define DHEAD 16
#define DEGC  16

typedef __attribute__((ext_vector_type(16))) _Float16 v16h;
typedef __attribute__((ext_vector_type(8)))  float    v8f;

// ---------------------------------------------------------------------------
// Kernel 1: QKV projection.  grid = (N1/16, 3), block = 256 (8 waves).
// blockIdx.y: 0 -> Q = x@wq, 1 -> K = x@wk, 2 -> V = x@wv.
// Each block stages a 16x128 f16 tile of x in LDS; each wave owns one 16-col
// output tile and chains 4 WMMAs over K=128.
// ---------------------------------------------------------------------------
__global__ __launch_bounds__(256) void qkv_proj_kernel(
    const float* __restrict__ x,
    const float* __restrict__ wq, const float* __restrict__ wk,
    const float* __restrict__ wv,
    float* __restrict__ qws, float* __restrict__ kws, float* __restrict__ vws)
{
    __shared__ _Float16 sA[16 * INDIM];   // 4 KB

    const int tid     = threadIdx.x;
    const int rowbase = blockIdx.x * 16;
    const float* w    = (blockIdx.y == 0) ? wq : (blockIdx.y == 1 ? wk : wv);
    float*       outp = (blockIdx.y == 0) ? qws : (blockIdx.y == 1 ? kws : vws);

    // Stage 16x128 fp32 -> f16 into LDS (2048 elems / 256 threads = 8 each).
    for (int i = tid; i < 16 * INDIM; i += 256) {
        const int r = i >> 7, c = i & 127;
        sA[i] = (_Float16)x[(size_t)(rowbase + r) * INDIM + c];
    }
    __syncthreads();

    const int lane = tid & 31;
    const int wav  = tid >> 5;          // 8 waves -> 8 column tiles of 16
    const int nloc = lane & 15;
    const int half = lane >> 4;
    const int col  = wav * 16 + nloc;   // global output column (0..127)

    v8f acc = {};
#pragma unroll
    for (int kk = 0; kk < 4; ++kk) {
        const int kbase = kk * 32;
        v16h a, b;
        // A (16x32 f16): lane = row M; halves 0..7 -> K koff..koff+7,
        // halves 8..15 -> K 16+koff..16+koff+7, koff = 8*(lane>=16).
        const int koff = half * 8;
#pragma unroll
        for (int i = 0; i < 8; ++i) {
            a[i]     = sA[nloc * INDIM + kbase + koff + i];
            a[i + 8] = sA[nloc * INDIM + kbase + 16 + koff + i];
        }
        // B (32x16 f16): lane = col N; same K order as A.
        const int kb = half * 8;
#pragma unroll
        for (int i = 0; i < 16; ++i)
            b[i] = (_Float16)w[(size_t)(kbase + kb + (i & 7) + (i >> 3) * 16) * HDIM + col];

        acc = __builtin_amdgcn_wmma_f32_16x16x32_f16(
                  false, a, false, b, (short)0, acc, false, false);
    }

    // C/D layout: lane holds column N=lane%16; VGPR r holds M = 8*(lane>=16)+r.
#pragma unroll
    for (int r = 0; r < 8; ++r) {
        const int m = half * 8 + r;
        outp[(size_t)(rowbase + m) * HDIM + col] = acc[r];
    }
}

// ---------------------------------------------------------------------------
// Kernel 2: fused per-node edge attention.  grid = N2, block = 256 (8 waves),
// wave h handles head h of node blockIdx.x.
//   e1 tile  = edge_attr_tile(16x128) @ we1[:, h*16:(h+1)*16]   (4 WMMAs)
//   score    = sum_d e1*k_n*q + (edge_attr @ we2)[:,h] + be2[h] (lane reduce)
//   softmax over deg=16, then attn-weighted v gather.
// e1 never leaves registers; edge_attr is read from HBM exactly once.
// ---------------------------------------------------------------------------
__global__ __launch_bounds__(256) void edge_attn_kernel(
    const float* __restrict__ ea,
    const float* __restrict__ we1, const float* __restrict__ we2,
    const float* __restrict__ be2,
    const long long* __restrict__ nbr_idx,   // edge_index row 0
    const float* __restrict__ qws, const float* __restrict__ kws,
    const float* __restrict__ vws,
    float* __restrict__ out)
{
    __shared__ _Float16 sEA[DEGC * INDIM];   // 4 KB: node's 16 edge rows, f16
    __shared__ int      sNbr[DEGC];

    const int       n     = blockIdx.x;
    const int       tid   = threadIdx.x;
    const long long ebase = (long long)n * DEGC;

    for (int i = tid; i < DEGC * INDIM; i += 256) {
        const int r = i >> 7, c = i & 127;
        sEA[i] = (_Float16)ea[(ebase + r) * INDIM + c];
    }
    if (tid < DEGC) sNbr[tid] = (int)nbr_idx[ebase + tid];
    __syncthreads();

    const int lane = tid & 31;
    const int h    = tid >> 5;        // head
    const int d    = lane & 15;       // D index (B/C column), also A row reuse
    const int half = lane >> 4;

    // ---- e1 tile via WMMA: A = edge_attr tile (M=deg, K), B = we1 cols ----
    v8f c = {};
#pragma unroll
    for (int kk = 0; kk < 4; ++kk) {
        const int kbase = kk * 32;
        v16h a, b;
        const int koff = half * 8;
#pragma unroll
        for (int i = 0; i < 8; ++i) {
            a[i]     = sEA[d * INDIM + kbase + koff + i];        // row M = lane%16
            a[i + 8] = sEA[d * INDIM + kbase + 16 + koff + i];
        }
        const int kb = half * 8;
#pragma unroll
        for (int i = 0; i < 16; ++i)
            b[i] = (_Float16)we1[(size_t)(kbase + kb + (i & 7) + (i >> 3) * 16) * HDIM + h * DHEAD + d];

        c = __builtin_amdgcn_wmma_f32_16x16x32_f16(
                false, a, false, b, (short)0, c, false, false);
    }
    // c[r] = e1[deg = half*8 + r][d]  for this head.

    // ---- scores: per-lane partial, then 16-lane reduction over d ----------
    const float qv = qws[(size_t)n * HDIM + h * DHEAD + d];

    float w2v[8];                                  // we2 k-slice for this lane
#pragma unroll
    for (int j = 0; j < 8; ++j)
        w2v[j] = we2[(size_t)(d * 8 + j) * HEADS + h];

    float sc[8];
#pragma unroll
    for (int r = 0; r < 8; ++r) {
        const int   deg = half * 8 + r;
        const int   nb  = sNbr[deg];
        const float kn  = kws[(size_t)nb * HDIM + h * DHEAD + d];
        float e2p = 0.f;                           // this lane's k-slice of e2
#pragma unroll
        for (int j = 0; j < 8; ++j)
            e2p += (float)sEA[deg * INDIM + d * 8 + j] * w2v[j];
        sc[r] = c[r] * kn * qv + e2p;
    }
    // Sum across the 16 lanes of this half (covers all d and all of e2's K).
#pragma unroll
    for (int m = 1; m <= 8; m <<= 1) {
#pragma unroll
        for (int r = 0; r < 8; ++r)
            sc[r] += __shfl_xor(sc[r], m, 32);
    }

    const float bias = be2[h];
    float mx = -1e30f;
#pragma unroll
    for (int r = 0; r < 8; ++r) {
        sc[r] = fminf(8.f, fmaxf(-8.f, sc[r] + bias));   // clip(-8, 8)
        mx = fmaxf(mx, sc[r]);
    }
    mx = fmaxf(mx, __shfl_xor(mx, 16, 32));              // max over both halves

    float se = 0.f;
#pragma unroll
    for (int r = 0; r < 8; ++r) { sc[r] = __expf(sc[r] - mx); se += sc[r]; }
    se += __shfl_xor(se, 16, 32);
    const float inv = 1.f / se;

    // ---- attn-weighted V gather ------------------------------------------
    float accv = 0.f;
#pragma unroll
    for (int r = 0; r < 8; ++r) {
        const int deg = half * 8 + r;
        const int nb  = sNbr[deg];
        accv += sc[r] * vws[(size_t)nb * HDIM + h * DHEAD + d];
    }
    accv *= inv;
    accv += __shfl_xor(accv, 16, 32);                    // combine deg halves
    if (half == 0)
        out[(size_t)n * HDIM + h * DHEAD + d] = accv;
}

// ---------------------------------------------------------------------------
extern "C" void kernel_launch(void* const* d_in, const int* in_sizes, int n_in,
                              void* d_out, int out_size, void* d_ws, size_t ws_size,
                              hipStream_t stream)
{
    const float*     x    = (const float*)d_in[0];
    const float*     ea   = (const float*)d_in[1];
    const float*     wq   = (const float*)d_in[2];
    const float*     wk   = (const float*)d_in[3];
    const float*     wv   = (const float*)d_in[4];
    const float*     we1  = (const float*)d_in[5];
    const float*     we2  = (const float*)d_in[6];
    const float*     be2  = (const float*)d_in[7];
    const long long* eidx = (const long long*)d_in[8];   // int64 [2, E]; row 0 = sources
    float*           out  = (float*)d_out;

    // Workspace: Q, K, V projections, fp32. 3 * 32768 * 128 * 4 B = 48 MiB.
    float* qws = (float*)d_ws;
    float* kws = qws + (size_t)N1C * HDIM;
    float* vws = kws + (size_t)N1C * HDIM;

    dim3 g1(N1C / 16, 3);
    qkv_proj_kernel<<<g1, 256, 0, stream>>>(x, wq, wk, wv, qws, kws, vws);

    edge_attn_kernel<<<N2C, 256, 0, stream>>>(ea, we1, we2, be2, eidx,
                                              qws, kws, vws, out);
}
